// MyModel_47622597378689
// MI455X (gfx1250) — compile-verified, matrix operand order checked by reading, not run on hardware
//
#include <hip/hip_runtime.h>
#include <math.h>

// ---------------- model constants ----------------
constexpr int Bn   = 64;    // batch (sentence pairs)
constexpr int Ln   = 128;   // nodes per graph
constexpr int Dd   = 300;   // hidden dim
constexpr int Hh   = 4;     // GAT heads
constexpr int HD   = Hh * Dd;   // 1200
constexpr int KP   = 320;   // K padded to multiple of 32 (10 wmma K-steps)
constexpr int NBP  = 320;   // Bt rows padded to multiple of 64 (for 300-col operands)
constexpr int NWP  = 1216;  // gatWt rows padded to multiple of 64 (for 1200 cols)
constexpr int NC   = 304;   // ldc padding for f32 C buffers with 300 real cols
constexpr float NEG_SLOPE = 0.2f;
constexpr float THRESH    = 0.1f;

// ---------------- WMMA types ----------------
typedef __attribute__((ext_vector_type(16))) _Float16 v16h;
typedef __attribute__((ext_vector_type(8)))  float    v8f;
typedef __attribute__((ext_vector_type(4)))  float    f32x4;

union TileU { v16h v; f32x4 q[2]; };

// Load one operand tile fragment per CDNA5 16-bit A/B layout:
// lane&15 = row within tile, lane>>4 selects K-half.
// VGPR0..3 hold K = khalf*8 + {0..7}, VGPR4..7 hold K = 16 + khalf*8 + {0..7}
__device__ __forceinline__ v16h load_tile16(const _Float16* rowptr, int k0, int khalf) {
  TileU u;
  const _Float16* p = rowptr + k0 + khalf * 8;
  u.q[0] = *(const f32x4*)(p);        // 8 contiguous halves (16B)
  u.q[1] = *(const f32x4*)(p + 16);   // next K-quarter (16B)
  return u.v;
}

__device__ __forceinline__ v8f wmma16(v16h a, v16h b, v8f c) {
  return __builtin_amdgcn_wmma_f32_16x16x32_f16(false, a, false, b,
                                                (short)0, c, false, false);
}

// ---------------- register-blocked strided-batched WMMA GEMM ----------------
// C[M=128, N] = A[M, K] * B[K, N], B supplied transposed: Bt[N][K].
// One wave per 2x4 block of 16x16 C tiles (32 rows x 64 cols):
//   6 operand-tile loads feed 8 WMMAs per K-step (vs 2:1 unblocked).
__global__ void k_gemm_wmma(const _Float16* __restrict__ A, size_t saB, int lda,
                            const _Float16* __restrict__ Bt, size_t sbB, int ldb,
                            float* __restrict__ C, size_t scB, int ldc,
                            int Nreal, int Ktiles)
{
  size_t bz = blockIdx.z;
  A  += bz * saB;  Bt += bz * sbB;  C += bz * scB;

  int lane  = threadIdx.x;
  int r15   = lane & 15;
  int khalf = lane >> 4;

  const _Float16* arow0 = A + (size_t)(blockIdx.y * 32 + r15) * lda;
  const _Float16* arow1 = arow0 + (size_t)16 * lda;
  const _Float16* brow[4];
  #pragma unroll
  for (int n = 0; n < 4; ++n)
    brow[n] = Bt + (size_t)(blockIdx.x * 64 + n * 16 + r15) * ldb;

  v8f c[2][4] = {};
  for (int kt = 0; kt < Ktiles; ++kt) {
    int k0 = kt * 32;
    v16h a0 = load_tile16(arow0, k0, khalf);
    v16h a1 = load_tile16(arow1, k0, khalf);
    #pragma unroll
    for (int n = 0; n < 4; ++n) {
      v16h b = load_tile16(brow[n], k0, khalf);
      c[0][n] = wmma16(a0, b, c[0][n]);
      c[1][n] = wmma16(a1, b, c[1][n]);
    }
  }
  // C/D layout: VGPR r -> row = khalf*8 + r, col = lane&15
  #pragma unroll
  for (int mi = 0; mi < 2; ++mi) {
    int mbase = blockIdx.y * 32 + mi * 16 + khalf * 8;
    #pragma unroll
    for (int n = 0; n < 4; ++n) {
      int col = blockIdx.x * 64 + n * 16 + r15;
      if (col < Nreal) {
        #pragma unroll
        for (int r = 0; r < 8; ++r)
          C[(size_t)(mbase + r) * ldc + col] = c[mi][n][r];
      }
    }
  }
}

// ---------------- adjacency via WMMA: G = V * V^T, threshold, +eye ----------------
// V: [B][128][KP] f16 (relu(h*simw), zero-padded). adj out: [B][128][128] f16 {0,1}
// Same 2x4 register blocking; grid (2, 4, B).
__global__ void k_adj_wmma(const _Float16* __restrict__ V, _Float16* __restrict__ adj)
{
  size_t b = blockIdx.z;
  const _Float16* Vb = V + b * (size_t)Ln * KP;

  int lane = threadIdx.x;
  int r15 = lane & 15, khalf = lane >> 4;

  const _Float16* arow0 = Vb + (size_t)(blockIdx.y * 32 + r15) * KP;
  const _Float16* arow1 = arow0 + (size_t)16 * KP;
  const _Float16* brow[4];
  #pragma unroll
  for (int n = 0; n < 4; ++n)
    brow[n] = Vb + (size_t)(blockIdx.x * 64 + n * 16 + r15) * KP;  // V^T col = V row

  v8f c[2][4] = {};
  for (int kt = 0; kt < KP / 32; ++kt) {
    int k0 = kt * 32;
    v16h a0 = load_tile16(arow0, k0, khalf);
    v16h a1 = load_tile16(arow1, k0, khalf);
    #pragma unroll
    for (int n = 0; n < 4; ++n) {
      v16h bt = load_tile16(brow[n], k0, khalf);
      c[0][n] = wmma16(a0, bt, c[0][n]);
      c[1][n] = wmma16(a1, bt, c[1][n]);
    }
  }
  _Float16* out = adj + b * (size_t)(Ln * Ln);
  #pragma unroll
  for (int mi = 0; mi < 2; ++mi) {
    int mbase = blockIdx.y * 32 + mi * 16 + khalf * 8;
    #pragma unroll
    for (int n = 0; n < 4; ++n) {
      int col = blockIdx.x * 64 + n * 16 + r15;
      #pragma unroll
      for (int r = 0; r < 8; ++r) {
        int m = mbase + r;
        float val = ((c[mi][n][r] >= THRESH) || (m == col)) ? 1.0f : 0.0f;
        out[(size_t)m * Ln + col] = (_Float16)val;
      }
    }
  }
}

// ---------------- elementwise / data-movement kernels ----------------

// h[b][l][d] = emb[sent[l*B+b]][d]   (sentence is [L,B])
__global__ void k_embed(const float* __restrict__ emb, const int* __restrict__ sent,
                        float* __restrict__ h)
{
  int bl = blockIdx.x;            // enumerates l*B + b
  int b = bl % Bn, l = bl / Bn;
  const float* src = emb + (size_t)sent[bl] * Dd;
  float* dst = h + ((size_t)b * Ln + l) * Dd;
  for (int d = threadIdx.x; d < Dd; d += blockDim.x) dst[d] = src[d];
}

// f16 cast with K-padding: dst[row][k<ncols ? src : 0], dst rows of KP
__global__ void k_castpad(const float* __restrict__ src, _Float16* __restrict__ dst,
                          int ncols, int total)
{
  for (int i = blockIdx.x * blockDim.x + threadIdx.x; i < total;
       i += gridDim.x * blockDim.x) {
    int k = i % KP, row = i / KP;
    float v = (k < ncols) ? src[(size_t)row * ncols + k] : 0.0f;
    dst[i] = (_Float16)v;
  }
}

// v = relu(h * simw), f16, K-padded
__global__ void k_vrelu(const float* __restrict__ h, const float* __restrict__ simw,
                        _Float16* __restrict__ v, int total)
{
  for (int i = blockIdx.x * blockDim.x + threadIdx.x; i < total;
       i += gridDim.x * blockDim.x) {
    int k = i % KP, row = i / KP;
    float x = 0.0f;
    if (k < Dd) { x = h[(size_t)row * Dd + k] * simw[k]; x = x > 0.0f ? x : 0.0f; }
    v[i] = (_Float16)x;
  }
}

// weight transpose to f16: Wt[n][k] (Npad rows, KP cols) from W[k][n]
__global__ void k_transW(const float* __restrict__ W, _Float16* __restrict__ Wt,
                         int K, int N, int Npad)
{
  int total = Npad * KP;
  for (int i = blockIdx.x * blockDim.x + threadIdx.x; i < total;
       i += gridDim.x * blockDim.x) {
    int k = i % KP, n = i / KP;
    float v = (k < K && n < N) ? W[(size_t)k * N + n] : 0.0f;
    Wt[i] = (_Float16)v;
  }
}

// ht[b][n][j] = f16(h[b][j][n]) : Bt operand of A@h  (NBP rows per batch)
__global__ void k_trans_h(const float* __restrict__ h, _Float16* __restrict__ ht)
{
  int total = Bn * NBP * Ln;
  for (int i = blockIdx.x * blockDim.x + threadIdx.x; i < total;
       i += gridDim.x * blockDim.x) {
    int j = i % Ln, n = (i / Ln) % NBP, b = i / (Ln * NBP);
    float v = (n < Dd) ? h[((size_t)b * Ln + j) * Dd + n] : 0.0f;
    ht[i] = (_Float16)v;
  }
}

// zt[(b*H+hh)][n][j] = f16(z[b][j][hh*300+n]) : Bt operand of alpha@z
__global__ void k_trans_z(const float* __restrict__ z, _Float16* __restrict__ zt)
{
  int total = Bn * Hh * NBP * Ln;
  for (int i = blockIdx.x * blockDim.x + threadIdx.x; i < total;
       i += gridDim.x * blockDim.x) {
    int j = i % Ln, n = (i / Ln) % NBP, bh = i / (Ln * NBP);
    int b = bh / Hh, hh = bh % Hh;
    float v = (n < Dd) ? z[((size_t)b * Ln + j) * HD + hh * Dd + n] : 0.0f;
    zt[i] = (_Float16)v;
  }
}

// el/er: per-(b,l,h) dot of z row with attn vectors. block=(32,4): wave per head.
__global__ void k_scores(const float* __restrict__ z, const float* __restrict__ attn_l,
                         const float* __restrict__ attn_r,
                         float* __restrict__ el, float* __restrict__ er)
{
  int bl = blockIdx.x;               // b*Ln + l
  int hh = threadIdx.y, lane = threadIdx.x;
  const float* zrow = z + (size_t)bl * HD + hh * Dd;
  float sl = 0.f, sr = 0.f;
  for (int d = lane; d < Dd; d += 32) {
    float zv = zrow[d];
    sl += zv * attn_l[hh * Dd + d];
    sr += zv * attn_r[hh * Dd + d];
  }
  for (int ofs = 16; ofs; ofs >>= 1) {
    sl += __shfl_down(sl, ofs, 32);
    sr += __shfl_down(sr, ofs, 32);
  }
  if (lane == 0) { el[bl * Hh + hh] = sl; er[bl * Hh + hh] = sr; }
}

// edge softmax over incoming edges (src j). block=(32,4): wave per head, 4 j per lane.
__global__ void k_softmax(const float* __restrict__ el, const float* __restrict__ er,
                          const _Float16* __restrict__ adj, _Float16* __restrict__ alpha)
{
  int bi = blockIdx.x;               // b*Ln + i (dst)
  int b = bi >> 7, i = bi & 127;
  int hh = threadIdx.y, lane = threadIdx.x;
  float eri = er[bi * Hh + hh];
  const _Float16* arow = adj + ((size_t)b * Ln + i) * Ln;

  float ev[4], mx = -1e30f;
  #pragma unroll
  for (int t = 0; t < 4; ++t) {
    int j = lane + 32 * t;
    float e = eri + el[(b * Ln + j) * Hh + hh];
    e = (e >= 0.f) ? e : NEG_SLOPE * e;               // leaky_relu
    e = ((float)arow[j] > 0.5f) ? e : -1e30f;         // edge mask
    ev[t] = e; mx = fmaxf(mx, e);
  }
  for (int ofs = 16; ofs; ofs >>= 1) mx = fmaxf(mx, __shfl_xor(mx, ofs, 32));
  float s = 0.f;
  #pragma unroll
  for (int t = 0; t < 4; ++t) { ev[t] = __expf(ev[t] - mx); s += ev[t]; }
  for (int ofs = 16; ofs; ofs >>= 1) s += __shfl_xor(s, ofs, 32);
  float inv = 1.0f / s;
  _Float16* out = alpha + ((size_t)(b * Hh + hh) * Ln + i) * Ln;
  #pragma unroll
  for (int t = 0; t < 4; ++t) out[lane + 32 * t] = (_Float16)(ev[t] * inv);
}

// deg[b][i] = row sum of SAGE adjacency (includes self loop)
__global__ void k_deg(const _Float16* __restrict__ adj, float* __restrict__ deg)
{
  int idx = blockIdx.x * blockDim.x + threadIdx.x;
  if (idx >= Bn * Ln) return;
  const _Float16* row = adj + (size_t)idx * Ln;
  float s = 0.f;
  for (int j = 0; j < Ln; ++j) s += (float)row[j];
  deg[idx] = s;
}

// neighb = f16((A@h + h) / (deg+1)), K-padded (A operand of sage GEMM)
__global__ void k_neigh(const float* __restrict__ nsum, const float* __restrict__ h,
                        const float* __restrict__ deg, _Float16* __restrict__ neighb,
                        int total)
{
  for (int i = blockIdx.x * blockDim.x + threadIdx.x; i < total;
       i += gridDim.x * blockDim.x) {
    int k = i % KP, row = i / KP;
    float v = 0.f;
    if (k < Dd)
      v = (nsum[(size_t)row * NC + k] + h[(size_t)row * Dd + k]) / (deg[row] + 1.0f);
    neighb[i] = (_Float16)v;
  }
}

// head fusion: sigmoid-gated sum of 4 GAT heads + SAGE head, residual into h.
// one wave per (b,i)
__global__ void k_fuse(float* __restrict__ hio,
                       const float* __restrict__ gat, const float* __restrict__ gat_b,
                       const float* __restrict__ sage, const float* __restrict__ sage_b,
                       const float* __restrict__ headW, const float* __restrict__ head_b)
{
  int bi = blockIdx.x;                 // b*Ln + i
  int b = bi >> 7, i = bi & 127;
  int lane = threadIdx.x;
  float w[5];
  #pragma unroll
  for (int head = 0; head < 5; ++head) {
    float dot = 0.f;
    for (int d = lane; d < Dd; d += 32) {
      float hv;
      if (head < 4)
        hv = gat[((size_t)(b * Hh + head) * Ln + i) * Dd + d] + gat_b[head * Dd + d];
      else
        hv = sage[((size_t)b * Ln + i) * NC + d] + sage_b[d];
      dot += hv * headW[d];
    }
    for (int ofs = 16; ofs; ofs >>= 1) dot += __shfl_xor(dot, ofs, 32);
    w[head] = 1.0f / (1.0f + __expf(-(dot + head_b[0])));
  }
  float* hrow = hio + (size_t)bi * Dd;
  for (int d = lane; d < Dd; d += 32) {
    float acc = 0.f;
    #pragma unroll
    for (int head = 0; head < 4; ++head)
      acc += w[head] * (gat[((size_t)(b * Hh + head) * Ln + i) * Dd + d] + gat_b[head * Dd + d]);
    acc += w[4] * (sage[((size_t)b * Ln + i) * NC + d] + sage_b[d]);
    hrow[d] += acc;                   // residual
  }
}

// WeightAndSum pool: per-node sigmoid gate (staged in LDS), sum over nodes.
__global__ void k_pool(const float* __restrict__ h, const float* __restrict__ poolW,
                       const float* __restrict__ pool_b, float* __restrict__ g)
{
  __shared__ float gate[Ln];
  int b = blockIdx.x, l = threadIdx.x;
  const float* hrow = h + ((size_t)b * Ln + l) * Dd;
  float dot = 0.f;
  for (int d = 0; d < Dd; ++d) dot += hrow[d] * poolW[d];
  gate[l] = 1.0f / (1.0f + __expf(-(dot + pool_b[0])));
  __syncthreads();
  for (int d = threadIdx.x; d < Dd; d += blockDim.x) {
    float s = 0.f;
    for (int l2 = 0; l2 < Ln; ++l2)
      s += h[((size_t)b * Ln + l2) * Dd + d] * gate[l2];
    g[(size_t)b * Dd + d] = s;
  }
}

// pearson * 5, one thread per batch element
__global__ void k_pearson(const float* __restrict__ g1, const float* __restrict__ g2,
                          float* __restrict__ out)
{
  int b = blockIdx.x * blockDim.x + threadIdx.x;
  if (b >= Bn) return;
  const float* a = g1 + (size_t)b * Dd;
  const float* c = g2 + (size_t)b * Dd;
  float m1 = 0.f, m2 = 0.f;
  for (int d = 0; d < Dd; ++d) { m1 += a[d]; m2 += c[d]; }
  m1 /= (float)Dd; m2 /= (float)Dd;
  float num = 0.f, s1 = 0.f, s2 = 0.f;
  for (int d = 0; d < Dd; ++d) {
    float x = a[d] - m1, y = c[d] - m2;
    num += x * y; s1 += x * x; s2 += y * y;
  }
  out[b] = 5.0f * num / (sqrtf(s1) * sqrtf(s2));
}

// ---------------- host-side orchestration ----------------
static inline char* carve(char*& p, size_t bytes) {
  char* r = p;
  p += (bytes + 255) & ~(size_t)255;
  return r;
}

struct LayerP {
  const float *gatW, *attn_l, *attn_r, *gat_b, *sageW, *sage_b, *headW, *head_b;
};

static inline void launch_gemm(hipStream_t s,
                               const _Float16* A, size_t saB, int lda,
                               const _Float16* Bt, size_t sbB, int ldb,
                               float* C, size_t scB, int ldc,
                               int Ngroups, int Nreal, int Ktiles, int batch)
{
  dim3 blk(32, 1, 1), grd(Ngroups, Ln / 32, batch);   // wave = 32x64 C block
  k_gemm_wmma<<<grd, blk, 0, s>>>(A, saB, lda, Bt, sbB, ldb, C, scB, ldc,
                                  Nreal, Ktiles);
}

extern "C" void kernel_launch(void* const* d_in, const int* in_sizes, int n_in,
                              void* d_out, int out_size, void* d_ws, size_t ws_size,
                              hipStream_t stream)
{
  (void)in_sizes; (void)n_in; (void)out_size; (void)ws_size;

  const float* emb    = (const float*)d_in[0];
  const float* simw1  = (const float*)d_in[1];
  const float* simw2  = (const float*)d_in[2];
  LayerP L[2];
  for (int ly = 0; ly < 2; ++ly) {
    int base = 3 + ly * 8;
    L[ly].gatW   = (const float*)d_in[base + 0];
    L[ly].attn_l = (const float*)d_in[base + 1];
    L[ly].attn_r = (const float*)d_in[base + 2];
    L[ly].gat_b  = (const float*)d_in[base + 3];
    L[ly].sageW  = (const float*)d_in[base + 4];
    L[ly].sage_b = (const float*)d_in[base + 5];
    L[ly].headW  = (const float*)d_in[base + 6];
    L[ly].head_b = (const float*)d_in[base + 7];
  }
  const float* poolW  = (const float*)d_in[19];
  const float* pool_b = (const float*)d_in[20];
  const int*   sent[2] = { (const int*)d_in[21], (const int*)d_in[22] };

  // ---- workspace carve ----
  char* p = (char*)d_ws;
  float*     hbuf[2];
  hbuf[0]          = (float*)    carve(p, sizeof(float)    * Bn * Ln * Dd);
  hbuf[1]          = (float*)    carve(p, sizeof(float)    * Bn * Ln * Dd);
  _Float16*  hb    = (_Float16*) carve(p, sizeof(_Float16) * Bn * Ln * KP);
  _Float16*  ht    = (_Float16*) carve(p, sizeof(_Float16) * Bn * NBP * Ln);
  _Float16*  vbuf  = (_Float16*) carve(p, sizeof(_Float16) * Bn * Ln * KP);
  _Float16*  adjg  = (_Float16*) carve(p, sizeof(_Float16) * Bn * Ln * Ln);
  _Float16*  adjs  = (_Float16*) carve(p, sizeof(_Float16) * Bn * Ln * Ln);
  float*     deg   = (float*)    carve(p, sizeof(float)    * Bn * Ln);
  float*     z     = (float*)    carve(p, sizeof(float)    * Bn * Ln * HD);
  _Float16*  zt    = (_Float16*) carve(p, sizeof(_Float16) * Bn * Hh * NBP * Ln);
  float*     el    = (float*)    carve(p, sizeof(float)    * Bn * Ln * Hh);
  float*     er    = (float*)    carve(p, sizeof(float)    * Bn * Ln * Hh);
  _Float16*  alpha = (_Float16*) carve(p, sizeof(_Float16) * Bn * Hh * Ln * Ln);
  float*     gat   = (float*)    carve(p, sizeof(float)    * Bn * Hh * Ln * Dd);
  float*     nsum  = (float*)    carve(p, sizeof(float)    * Bn * Ln * NC);
  _Float16*  neighb= (_Float16*) carve(p, sizeof(_Float16) * Bn * Ln * KP);
  float*     sage  = (float*)    carve(p, sizeof(float)    * Bn * Ln * NC);
  _Float16*  gatWt = (_Float16*) carve(p, sizeof(_Float16) * NWP * KP);
  _Float16*  sageWt= (_Float16*) carve(p, sizeof(_Float16) * NBP * KP);
  float*     gpool[2];
  gpool[0]         = (float*)    carve(p, sizeof(float)    * Bn * Dd);
  gpool[1]         = (float*)    carve(p, sizeof(float)    * Bn * Dd);

  const int EW = 256;
  auto blocks = [](int total, int t) { return (total + t - 1) / t; };
  const int totPad = Bn * Ln * KP;

  for (int br = 0; br < 2; ++br) {
    float* h = hbuf[br];
    // embedding gather with [L,B] -> [B,L,D] transpose
    k_embed<<<Ln * Bn, 128, 0, stream>>>(emb, sent[br], h);

    for (int ly = 0; ly < 2; ++ly) {
      const LayerP& P = L[ly];
      // weight transposes (Bt operands), f16, K- and N-padded
      k_transW<<<blocks(NWP * KP, EW), EW, 0, stream>>>(P.gatW, gatWt, Dd, HD, NWP);
      k_transW<<<blocks(NBP * KP, EW), EW, 0, stream>>>(P.sageW, sageWt, Dd, Dd, NBP);
      // f16 views of h
      k_castpad<<<blocks(totPad, EW), EW, 0, stream>>>(h, hb, Dd, totPad);
      k_trans_h<<<blocks(Bn * NBP * Ln, EW), EW, 0, stream>>>(h, ht);

      // --- graph build (WMMA V*V^T, threshold, +self-loops) ---
      k_vrelu<<<blocks(totPad, EW), EW, 0, stream>>>(h, simw2, vbuf, totPad);
      { dim3 blk(32, 1, 1), grd(2, 4, Bn);
        k_adj_wmma<<<grd, blk, 0, stream>>>(vbuf, adjg); }     // GAT graph
      k_vrelu<<<blocks(totPad, EW), EW, 0, stream>>>(h, simw1, vbuf, totPad);
      { dim3 blk(32, 1, 1), grd(2, 4, Bn);
        k_adj_wmma<<<grd, blk, 0, stream>>>(vbuf, adjs); }     // SAGE graph
      k_deg<<<blocks(Bn * Ln, EW), EW, 0, stream>>>(adjs, deg);

      // --- GAT: z = h @ gatW  (64 x [128x300]*[300x1200]) ---
      launch_gemm(stream, hb, (size_t)Ln * KP, KP,
                  gatWt, 0, KP,
                  z, (size_t)Ln * HD, HD,
                  NWP / 64, HD, KP / 32, Bn);
      k_scores<<<Bn * Ln, dim3(32, 4), 0, stream>>>(z, P.attn_l, P.attn_r, el, er);
      k_trans_z<<<blocks(Bn * Hh * NBP * Ln, EW), EW, 0, stream>>>(z, zt);
      k_softmax<<<Bn * Ln, dim3(32, 4), 0, stream>>>(el, er, adjg, alpha);
      // gat = alpha @ z  (256 x [128x128]*[128x300])
      launch_gemm(stream, alpha, (size_t)Ln * Ln, Ln,
                  zt, (size_t)NBP * Ln, Ln,
                  gat, (size_t)Ln * Dd, Dd,
                  NBP / 64, Dd, Ln / 32, Bn * Hh);

      // --- SAGE (gcn): nsum = A @ h, neigh = (nsum+h)/(deg+1), sage = neigh @ W ---
      launch_gemm(stream, adjs, (size_t)Ln * Ln, Ln,
                  ht, (size_t)NBP * Ln, Ln,
                  nsum, (size_t)Ln * NC, NC,
                  NBP / 64, Dd, Ln / 32, Bn);
      k_neigh<<<blocks(totPad, EW), EW, 0, stream>>>(nsum, h, deg, neighb, totPad);
      launch_gemm(stream, neighb, (size_t)Ln * KP, KP,
                  sageWt, 0, KP,
                  sage, (size_t)Ln * NC, NC,
                  NBP / 64, Dd, KP / 32, Bn);

      // --- head fusion + residual (in place on h) ---
      k_fuse<<<Bn * Ln, 32, 0, stream>>>(h, gat, P.gat_b, sage, P.sage_b,
                                         P.headW, P.head_b);
    }
    k_pool<<<Bn, Ln, 0, stream>>>(h, poolW, pool_b, gpool[br]);
  }
  k_pearson<<<1, Bn, 0, stream>>>(gpool[0], gpool[1], (float*)d_out);
}
